// SimpleRoPEAttention_16346645529113
// MI455X (gfx1250) — matrix operand order repaired and checked by reading, not compiled, not run on hardware
//
#include <hip/hip_runtime.h>
#include <hip/hip_bf16.h>

// ---------- vector types ----------
typedef __attribute__((ext_vector_type(16))) __bf16 v16bf;
typedef __attribute__((ext_vector_type(8)))  __bf16 bf16x8;
typedef __attribute__((ext_vector_type(4)))  __bf16 bf16x4;
typedef __attribute__((ext_vector_type(8)))  float  v8f;
typedef __attribute__((ext_vector_type(4)))  float  f32x4;
typedef __attribute__((ext_vector_type(4)))  unsigned u32x4;
typedef __attribute__((ext_vector_type(8)))  unsigned u32x8;

// ---------- problem constants ----------
static constexpr int BATCH = 4;
static constexpr int SEQ   = 2048;
static constexpr int DMOD  = 1024;
static constexpr int NHEAD = 16;
static constexpr int HDIM  = 64;
static constexpr int MROWS = BATCH * SEQ;   // 8192

// ---------------------------------------------------------------------------
// CDNA5 async copy: global -> LDS, 16B per lane, tracked by ASYNCcnt.
// LDS destination address = low 32 bits of the generic shared pointer
// (flat LDS-aperture addresses carry the wave-relative LDS offset, ISA 10.2).
// ---------------------------------------------------------------------------
__device__ __forceinline__ void async_copy16(void* lds_dst, const void* gsrc) {
  unsigned ldsa = (unsigned)(unsigned long long)lds_dst;
  asm volatile("global_load_async_to_lds_b128 %0, %1, off"
               :: "v"(ldsa), "v"(gsrc)
               : "memory");
}
__device__ __forceinline__ void async_wait0() {
  asm volatile("s_wait_asynccnt 0" ::: "memory");
}

// ---------------------------------------------------------------------------
// CDNA5 Tensor Data Mover: DMA a 2-D tile (tile_w x tile_h elements of 2B)
// from global memory into LDS, with hardware row padding so the LDS row
// stride stays bank-conflict-free. Issued by one wave; tracked by TENSORcnt.
// D# layout per CDNA5 ISA (group0 128b + group1 256b).
// ---------------------------------------------------------------------------
__device__ __forceinline__ void tdm_load_2d(void* lds_dst, const void* gsrc,
                                            unsigned tile_w, unsigned tile_h,
                                            unsigned row_stride_elems,
                                            unsigned pad_interval_code,
                                            unsigned pad_amount_code) {
  unsigned long long ga = (unsigned long long)gsrc;
  u32x4 g0;
  g0[0] = 1u;                                         // count=1, user mode
  g0[1] = (unsigned)(unsigned long long)lds_dst;      // lds_addr (bytes)
  g0[2] = (unsigned)ga;                               // global_addr[31:0]
  g0[3] = (unsigned)((ga >> 32) & 0x01FFFFFFu)        // global_addr[56:32]
        | 0x80000000u;                                // type=2 ("image")
  const unsigned td0 = row_stride_elems;              // tensor_dim0 (no OOB)
  const unsigned td1 = 0x40000000u;                   // tensor_dim1 (huge)
  u32x8 g1;
  g1[0] = (1u << 16)                                  // data_size = 2 bytes
        | (1u << 20)                                  // pad_enable
        | (pad_interval_code << 22)
        | (pad_amount_code << 25);
  g1[1] = (td0 & 0xFFFFu) << 16;                      // tensor_dim0 lo
  g1[2] = (td0 >> 16) | ((td1 & 0xFFFFu) << 16);      // dim0 hi | dim1 lo
  g1[3] = (td1 >> 16) | (tile_w << 16);               // dim1 hi | tile_dim0
  g1[4] = tile_h & 0xFFFFu;                           // tile_dim1 (dim2=0)
  g1[5] = row_stride_elems;                           // tensor_dim0_stride lo
  g1[6] = 0u;                                         // stride hi | dim1_stride
  g1[7] = 0u;
  asm volatile("tensor_load_to_lds %0, %1" :: "s"(g0), "s"(g1) : "memory");
}
__device__ __forceinline__ void tensor_wait0() {
  __builtin_amdgcn_s_wait_tensorcnt((short)0);
}

// ---------------------------------------------------------------------------
// WMMA helpers (CDNA5 gfx1250: D = A(16x32 bf16) * B(32x16 bf16) + C(16x16 f32))
// ---------------------------------------------------------------------------
__device__ __forceinline__ v8f wmma_bf16(v16bf a, v16bf b, v8f c) {
  return __builtin_amdgcn_wmma_f32_16x16x32_bf16(
      /*neg_a=*/false, a, /*neg_b=*/false, b,
      /*c_mod=*/(short)0, c, /*reuse_a=*/false, /*reuse_b=*/false);
}

// A fragment (16x32, M x K): lane m = L&15; lo = (L<16)?0:8,
// half i -> K = (i>>3)*16 + lo + (i&7)  => two contiguous 16B LDS loads.
__device__ __forceinline__ v16bf load_frag_a(const __bf16* lds, int stride,
                                             int row0, int k0) {
  const int lane = threadIdx.x & 31;
  const int r    = row0 + (lane & 15);
  const int lo   = (lane < 16) ? 0 : 8;
  union { v16bf v; bf16x8 h[2]; } u;
  u.h[0] = *(const bf16x8*)(lds + r * stride + k0 + lo);
  u.h[1] = *(const bf16x8*)(lds + r * stride + k0 + 16 + lo);
  return u.v;
}

// B fragment (32x16, K x N) from LDS stored [N][K]: lane n = L&15;
// half i -> K = (i>>3)*16 + 8*(L>=16) + (i&7)  => two 16B loads.
__device__ __forceinline__ v16bf load_frag_b(const __bf16* lds, int stride,
                                             int col0, int k0) {
  const int lane = threadIdx.x & 31;
  const int c    = col0 + (lane & 15);
  const int ko   = k0 + ((lane >> 4) << 3);
  union { v16bf v; bf16x8 h[2]; } u;
  u.h[0] = *(const bf16x8*)(lds + c * stride + ko);
  u.h[1] = *(const bf16x8*)(lds + c * stride + ko + 16);
  return u.v;
}

// ---------------------------------------------------------------------------
// fp32 -> bf16 weight conversion
// ---------------------------------------------------------------------------
__global__ __launch_bounds__(256)
void cvt_bf16_kernel(const float* __restrict__ src, __bf16* __restrict__ dst,
                     int n4) {
  int i = blockIdx.x * 256 + threadIdx.x;
  if (i < n4) {
    f32x4 v = *(const f32x4*)(src + (size_t)i * 4);
    bf16x4 o;
#pragma unroll
    for (int e = 0; e < 4; ++e) o[e] = (__bf16)v[e];
    *(bf16x4*)(dst + (size_t)i * 4) = o;
  }
}

// ---------------------------------------------------------------------------
// RoPE: rx = rotate_half(x) over full feature dim, fp32 in -> bf16 out
// ---------------------------------------------------------------------------
__global__ __launch_bounds__(256)
void rope_kernel(const float* __restrict__ x, __bf16* __restrict__ rx) {
  const int row = blockIdx.x;        // 0..8191
  const int s   = row & (SEQ - 1);   // position within sequence
  const float* xr = x + (size_t)row * DMOD;
  __bf16*      o  = rx + (size_t)row * DMOD;
  const int half = DMOD / 2;         // 512
  for (int i = threadIdx.x; i < half; i += 256) {
    float freq = __expf(-9.210340371976184f * (float)(2 * i) * (1.0f / (float)DMOD));
    float ang  = (float)s * freq;
    float sn, cs;
    __sincosf(ang, &sn, &cs);
    float x1 = xr[i], x2 = xr[i + half];
    o[i]        = (__bf16)(x1 * cs - x2 * sn);
    o[i + half] = (__bf16)(x1 * sn + x2 * cs);
  }
}

// ---------------------------------------------------------------------------
// GEMM: C[M,N] = A[M,K](bf16) * W[N,K]^T(bf16) + bias(f32)
// 128x128x32 block tile, 8 waves (4Mx2N), each wave 32x64 = 2x4 WMMA tiles.
// Double-buffered LDS tiles filled by the Tensor Data Mover (one descriptor
// per tile, issued by wave 0; TDM pads each 64B row by 16B -> LDT=40 halfs).
// One barrier per k-step; the DMA overlaps the WMMAs.
// ---------------------------------------------------------------------------
template <bool OUT_BF16>
__global__ __launch_bounds__(256)
void gemm_bias_kernel(const __bf16* __restrict__ A, const __bf16* __restrict__ W,
                      const float* __restrict__ bias, void* __restrict__ out,
                      int M, int N, int K) {
  constexpr int BM = 128, BN = 128, BK = 32, LDT = BK + 8;  // padded stride
  __shared__ __bf16 As[2][BM * LDT];
  __shared__ __bf16 Ws[2][BN * LDT];

  const int tid  = threadIdx.x;
  const int wave = tid >> 5;
  const int wm   = wave >> 1;       // 0..3 -> 32-row group
  const int wn   = wave & 1;        // 0..1 -> 64-col group
  const int m0   = blockIdx.x * BM;
  const int n0   = blockIdx.y * BN;

  // TDM pad codes: 16 DWORDs (64B row) between pads, 4 DWORDs (16B) of pad.
  auto issue_tiles = [&](int k0, int buf) {
    if (wave == 0) {
      tdm_load_2d(&As[buf][0], &A[(size_t)m0 * K + k0], BK, BM, K, 3, 3);
      tdm_load_2d(&Ws[buf][0], &W[(size_t)n0 * K + k0], BK, BN, K, 3, 3);
    }
  };

  v8f acc[2][4];
#pragma unroll
  for (int s = 0; s < 2; ++s)
#pragma unroll
    for (int t = 0; t < 4; ++t) {
      v8f z = {0.f, 0.f, 0.f, 0.f, 0.f, 0.f, 0.f, 0.f};
      acc[s][t] = z;
    }

  issue_tiles(0, 0);
  tensor_wait0();
  __syncthreads();

  int buf = 0;
  for (int k0 = 0; k0 < K; k0 += BK, buf ^= 1) {
    if (k0 + BK < K) issue_tiles(k0 + BK, buf ^ 1);  // DMA overlaps WMMAs

    v16bf a0 = load_frag_a(As[buf], LDT, wm * 32, 0);
    v16bf a1 = load_frag_a(As[buf], LDT, wm * 32 + 16, 0);
#pragma unroll
    for (int t = 0; t < 4; ++t) {
      v16bf b = load_frag_b(Ws[buf], LDT, wn * 64 + t * 16, 0);
      acc[0][t] = wmma_bf16(a0, b, acc[0][t]);
      acc[1][t] = wmma_bf16(a1, b, acc[1][t]);
    }
    tensor_wait0();
    __syncthreads();
  }

  // epilogue: C/D layout -> lane n = L&15, reg r -> m = r + 8*(L>=16)
  const int lane = tid & 31;
  const int cb   = lane & 15;
  const int rb   = (lane >> 4) * 8;
#pragma unroll
  for (int s = 0; s < 2; ++s)
#pragma unroll
    for (int t = 0; t < 4; ++t) {
      int col = n0 + wn * 64 + t * 16 + cb;
      float bval = bias[col];
#pragma unroll
      for (int r = 0; r < 8; ++r) {
        int row = m0 + wm * 32 + s * 16 + rb + r;
        float v = acc[s][t][r] + bval;
        if constexpr (OUT_BF16)
          ((__bf16*)out)[(size_t)row * N + col] = (__bf16)v;
        else
          ((float*)out)[(size_t)row * N + col] = v;
      }
    }
}

// ---------------------------------------------------------------------------
// Flash attention: one block = 128 queries of one (b,h); 8 waves x 16 queries.
// Streaming 64-key blocks, online softmax, all matmuls via WMMA bf16.
// Q tile: per-lane async copy engine (ASYNCcnt). K tile: Tensor Data Mover
// (TENSORcnt, 144B padded rows -> LD=72 halfs). V tile: VGPR path (transpose).
// ---------------------------------------------------------------------------
__global__ __launch_bounds__(256)
void flash_attn_kernel(const __bf16* __restrict__ Qb, const __bf16* __restrict__ Kb,
                       const __bf16* __restrict__ Vb, __bf16* __restrict__ Ob) {
  constexpr int LD  = HDIM + 8;   // 72-half padded stride
  constexpr int BQ  = 128;
  constexpr int BKV = 64;
  __shared__ __bf16 Qs[BQ * LD];        // [query][d]
  __shared__ __bf16 Ks[BKV * LD];       // [key][d]
  __shared__ __bf16 Vt[HDIM * LD];      // transposed: [d][key]
  __shared__ __bf16 Ps[8 * 16 * LD];    // per-wave P staging [q][key]

  const int tid  = threadIdx.x;
  const int lane = tid & 31;
  const int wave = tid >> 5;
  const int qblk = blockIdx.x;          // 0..15
  const int bh   = blockIdx.y;          // 0..63
  const int b    = bh >> 4;
  const int h    = bh & 15;

  const size_t rowbase = (size_t)b * SEQ;   // row offset in 8192-row matrices
  const int    colbase = h * HDIM;

  // ---- load Q tile (128 x 64) via async copies ----
#pragma unroll
  for (int j = 0; j < 4; ++j) {
    int c   = tid + j * 256;            // 0..1023 (8 chunks/row)
    int row = c >> 3;
    int kc  = (c & 7) * 8;
    async_copy16(&Qs[row * LD + kc],
                 &Qb[(rowbase + qblk * BQ + row) * DMOD + colbase + kc]);
  }
  async_wait0();
  __syncthreads();

  const v16bf qf0 = load_frag_a(Qs, LD, wave * 16, 0);
  const v16bf qf1 = load_frag_a(Qs, LD, wave * 16, 32);

  v8f o[4];
  float mrow[8], lrow[8];
#pragma unroll
  for (int t = 0; t < 4; ++t) {
    v8f z = {0.f, 0.f, 0.f, 0.f, 0.f, 0.f, 0.f, 0.f};
    o[t] = z;
  }
#pragma unroll
  for (int r = 0; r < 8; ++r) { mrow[r] = -1e30f; lrow[r] = 0.f; }

  const float sc = 0.125f;    // 1/sqrt(64)
  const int nkb = SEQ / BKV;  // 32 key blocks

  for (int kb = 0; kb < nkb; ++kb) {
    __syncthreads();  // previous iteration done with Ks/Vt
    // ---- K tile: TDM descriptor (wave 0); rows 128B + 16B pad -> 72 halfs ----
    if (wave == 0)
      tdm_load_2d(&Ks[0], &Kb[(rowbase + (size_t)kb * BKV) * DMOD + colbase],
                  HDIM, BKV, DMOD, 4, 3);
    // ---- V tile: VGPR load + transposed LDS store ----
#pragma unroll
    for (int j = 0; j < 2; ++j) {
      int c   = tid + j * 256;          // 0..511
      int key = c >> 3;
      int kc  = (c & 7) * 8;
      size_t g = (rowbase + kb * BKV + key) * (size_t)DMOD + colbase + kc;
      bf16x8 vv = *(const bf16x8*)&Vb[g];
#pragma unroll
      for (int e = 0; e < 8; ++e) Vt[(kc + e) * LD + key] = vv[e];
      if (kb + 1 < nkb)
        __builtin_prefetch(&Vb[g + (size_t)BKV * DMOD], 0, 1);
    }
    tensor_wait0();
    __syncthreads();

    // ---- scores: S(16x64) = Q(16x64) @ K^T ----
    v8f sf[4];
#pragma unroll
    for (int t = 0; t < 4; ++t) {
      v8f z = {0.f, 0.f, 0.f, 0.f, 0.f, 0.f, 0.f, 0.f};
      v16bf bk0 = load_frag_b(Ks, LD, t * 16, 0);
      v16bf bk1 = load_frag_b(Ks, LD, t * 16, 32);
      z = wmma_bf16(qf0, bk0, z);
      z = wmma_bf16(qf1, bk1, z);
      sf[t] = z;
    }
#pragma unroll
    for (int t = 0; t < 4; ++t)
#pragma unroll
      for (int r = 0; r < 8; ++r) sf[t][r] *= sc;

    // ---- online softmax (row = r + 8*(lane>=16); 16 lanes per row) ----
    float alpha[8];
#pragma unroll
    for (int r = 0; r < 8; ++r) {
      float mx = fmaxf(fmaxf(sf[0][r], sf[1][r]), fmaxf(sf[2][r], sf[3][r]));
#pragma unroll
      for (int off = 1; off < 16; off <<= 1)
        mx = fmaxf(mx, __shfl_xor(mx, off, 32));
      float mn = fmaxf(mrow[r], mx);
      alpha[r] = __expf(mrow[r] - mn);
      mrow[r]  = mn;
    }
    float rs[8];
#pragma unroll
    for (int r = 0; r < 8; ++r) rs[r] = 0.f;
#pragma unroll
    for (int t = 0; t < 4; ++t)
#pragma unroll
      for (int r = 0; r < 8; ++r) {
        float p = __expf(sf[t][r] - mrow[r]);
        sf[t][r] = p;
        rs[r] += p;
      }
#pragma unroll
    for (int r = 0; r < 8; ++r) {
#pragma unroll
      for (int off = 1; off < 16; off <<= 1)
        rs[r] += __shfl_xor(rs[r], off, 32);
      lrow[r] = lrow[r] * alpha[r] + rs[r];
    }

    // ---- stage P (16x64 bf16) into per-wave LDS, reload as A fragments ----
    __bf16* Pw = &Ps[wave * 16 * LD];
    const int crow = (lane >> 4) * 8;
    const int ccol = lane & 15;
#pragma unroll
    for (int t = 0; t < 4; ++t)
#pragma unroll
      for (int r = 0; r < 8; ++r)
        Pw[(crow + r) * LD + t * 16 + ccol] = (__bf16)sf[t][r];

    v16bf p0 = load_frag_a(Pw, LD, 0, 0);
    v16bf p1 = load_frag_a(Pw, LD, 0, 32);

    // ---- O = alpha*O + P @ V ----
#pragma unroll
    for (int t = 0; t < 4; ++t) {
#pragma unroll
      for (int r = 0; r < 8; ++r) o[t][r] *= alpha[r];
      o[t] = wmma_bf16(p0, load_frag_b(Vt, LD, t * 16, 0), o[t]);
      o[t] = wmma_bf16(p1, load_frag_b(Vt, LD, t * 16, 32), o[t]);
    }
  }

  // ---- normalize and write attended (bf16, (B,S,H*hd) row-major) ----
  const int crow = (lane >> 4) * 8;
  const int ccol = lane & 15;
#pragma unroll
  for (int r = 0; r < 8; ++r) {
    float inv = 1.0f / lrow[r];
    size_t row = rowbase + (size_t)qblk * BQ + wave * 16 + crow + r;
#pragma unroll
    for (int t = 0; t < 4; ++t)
      Ob[row * DMOD + colbase + t * 16 + ccol] = (__bf16)(o[t][r] * inv);
  }
}

// ---------------------------------------------------------------------------
// launch
// ---------------------------------------------------------------------------
extern "C" void kernel_launch(void* const* d_in, const int* in_sizes, int n_in,
                              void* d_out, int out_size, void* d_ws, size_t ws_size,
                              hipStream_t stream) {
  const float* x  = (const float*)d_in[0];
  const float* Wq = (const float*)d_in[1];
  const float* bq = (const float*)d_in[2];
  const float* Wk = (const float*)d_in[3];
  const float* bk = (const float*)d_in[4];
  const float* Wv = (const float*)d_in[5];
  const float* bv = (const float*)d_in[6];
  const float* Wo = (const float*)d_in[7];
  const float* bo = (const float*)d_in[8];

  char* p = (char*)d_ws;
  auto alloc = [&](size_t bytes) -> void* {
    void* r = (void*)p;
    p += (bytes + 255) & ~(size_t)255;
    return r;
  };
  const size_t act_bytes = (size_t)MROWS * DMOD * 2;  // 16 MB bf16
  const size_t w_bytes   = (size_t)DMOD * DMOD * 2;   // 2 MB bf16

  __bf16* rx  = (__bf16*)alloc(act_bytes);
  __bf16* wqb = (__bf16*)alloc(w_bytes);
  __bf16* wkb = (__bf16*)alloc(w_bytes);
  __bf16* wvb = (__bf16*)alloc(w_bytes);
  __bf16* wob = (__bf16*)alloc(w_bytes);
  __bf16* Qb  = (__bf16*)alloc(act_bytes);
  __bf16* Kb  = (__bf16*)alloc(act_bytes);
  __bf16* Vb  = (__bf16*)alloc(act_bytes);
  __bf16* att = (__bf16*)alloc(act_bytes);

  const int n4 = DMOD * DMOD / 4;  // 262144
  cvt_bf16_kernel<<<(n4 + 255) / 256, 256, 0, stream>>>(Wq, wqb, n4);
  cvt_bf16_kernel<<<(n4 + 255) / 256, 256, 0, stream>>>(Wk, wkb, n4);
  cvt_bf16_kernel<<<(n4 + 255) / 256, 256, 0, stream>>>(Wv, wvb, n4);
  cvt_bf16_kernel<<<(n4 + 255) / 256, 256, 0, stream>>>(Wo, wob, n4);

  rope_kernel<<<MROWS, 256, 0, stream>>>(x, rx);

  dim3 ggemm(MROWS / 128, DMOD / 128);  // 64 x 8
  gemm_bias_kernel<true><<<ggemm, 256, 0, stream>>>(rx, wqb, bq, (void*)Qb,
                                                    MROWS, DMOD, DMOD);
  gemm_bias_kernel<true><<<ggemm, 256, 0, stream>>>(rx, wkb, bk, (void*)Kb,
                                                    MROWS, DMOD, DMOD);
  gemm_bias_kernel<true><<<ggemm, 256, 0, stream>>>(rx, wvb, bv, (void*)Vb,
                                                    MROWS, DMOD, DMOD);

  dim3 gfa(SEQ / 128, BATCH * NHEAD);   // 16 x 64
  flash_attn_kernel<<<gfa, 256, 0, stream>>>(Qb, Kb, Vb, att);

  gemm_bias_kernel<false><<<ggemm, 256, 0, stream>>>(att, wob, bo, d_out,
                                                     MROWS, DMOD, DMOD);
}
